// MHSA_1382979470159
// MI455X (gfx1250) — matrix operand order repaired and checked by reading, not compiled, not run on hardware
//
#include <hip/hip_runtime.h>

// ---------------------------------------------------------------------------
// MI455X (gfx1250, wave32) rotary MHSA: bf16 WMMA (f32 accumulate) everywhere,
// GEMM LDS tiles staged by the Tensor Data Mover (double-buffered, TENSORcnt).
// ---------------------------------------------------------------------------

typedef __attribute__((ext_vector_type(16))) __bf16 v16bf;
typedef __attribute__((ext_vector_type(8)))  float  v8f;
typedef __attribute__((ext_vector_type(4)))  unsigned int v4u;
typedef __attribute__((ext_vector_type(8)))  int    v8i;
typedef __attribute__((ext_vector_type(4)))  int    v4i;

static constexpr int   BQ    = 64;
static constexpr int   NSEQ  = 197;
static constexpr int   DIM   = 768;
static constexpr int   HEADS = 12;
static constexpr int   DH    = 64;
static constexpr int   MROWS = BQ * NSEQ;   // 12608
static constexpr int   NQKV  = 3 * DIM;     // 2304
static constexpr float MASK_FILL = -987654321.0f;
static constexpr float NEG_BIG   = -3.0e38f;

#ifndef __has_builtin
#define __has_builtin(x) 0
#endif
#if __has_builtin(__builtin_amdgcn_tensor_load_to_lds) && \
    __has_builtin(__builtin_amdgcn_s_wait_tensorcnt)
#define HAVE_TDM 1
#else
#define HAVE_TDM 0
#endif
#if __has_include(<hip/amd_detail/amd_gfx1250_TDM.h>)
#define TDM_SIX_ARGS 1   // therock-10.0 headers => 6-arg builtin
#else
#define TDM_SIX_ARGS 0   // ROCm 7.2 => 5-arg builtin
#endif

__device__ __forceinline__ __bf16 f2bf(float f) {
  unsigned u = __float_as_uint(f);
  unsigned r = u + 0x7FFFu + ((u >> 16) & 1u);   // round-to-nearest-even
  union { unsigned short s; __bf16 b; } o;
  o.s = (unsigned short)(r >> 16);
  return o.b;
}

union Frag { v16bf v; uint4 u[2]; };   // 32 bytes: one WMMA A/B operand per lane

__device__ __forceinline__ v8f wmma_bf16(const v16bf& a, const v16bf& b, const v8f& c) {
  return __builtin_amdgcn_wmma_f32_16x16x32_bf16(
      /*neg_a=*/false, a, /*neg_b=*/false, b,
      /*c_mod=*/(short)0, c, /*reuse_a=*/false, /*reuse_b=*/false);
}

#if HAVE_TDM
// TDM 2-D tile load: global (row-major, element stride `stride`, 2-byte bf16)
// -> LDS, with 4-DWORD padding appended after every 16 DWORDs (64B row -> 80B
// LDS row = the [..][40] bf16 layout). OOB rows (>= td1) read as zero.
// Descriptor packing per CDNA5 ISA.
__device__ __forceinline__ void tdm_load_2d(unsigned lds_off, const void* gptr,
                                            unsigned td0, unsigned td1,
                                            unsigned tile0, unsigned tile1,
                                            unsigned stride) {
  unsigned long long ga = (unsigned long long)(size_t)gptr;
  v4u g0;
  g0[0] = 1u;                                    // count=1 (valid), user mode
  g0[1] = lds_off;                               // lds_addr (bytes)
  g0[2] = (unsigned)ga;                          // global_addr[31:0]
  g0[3] = (unsigned)(ga >> 32) | (2u << 30);     // global_addr[56:32] | type=2
  v8i g1;
  g1[0] = (int)((1u << 16)      // data_size = 1 -> 2 bytes (bf16)
              | (1u << 20)      // pad_enable
              | (3u << 22)      // pad_interval: 16 DWORDs (64B) of data...
              | (3u << 25));    // pad_amount: ...then 4 DWORDs (16B) of pad
  g1[1] = (int)((td0 & 0xFFFFu) << 16);                          // tensor_dim0 lo
  g1[2] = (int)(((td0 >> 16) & 0xFFFFu) | ((td1 & 0xFFFFu) << 16)); // td0 hi|td1 lo
  g1[3] = (int)(((td1 >> 16) & 0xFFFFu) | (tile0 << 16));        // td1 hi|tile_dim0
  g1[4] = (int)tile1;                                            // tile_dim1 (tile_dim2=0)
  g1[5] = (int)stride;                                           // tensor_dim0_stride lo
  g1[6] = 0;                                                     // stride hi | dim1_stride
  g1[7] = 0;
  v4i z4 = {0, 0, 0, 0};
#if TDM_SIX_ARGS
  v8i z8 = {0, 0, 0, 0, 0, 0, 0, 0};
  __builtin_amdgcn_tensor_load_to_lds(g0, g1, z4, z4, z8, 0);
#else
  __builtin_amdgcn_tensor_load_to_lds(g0, g1, z4, z4, 0);
#endif
}
#endif  // HAVE_TDM

// --------------------------- conversion kernels ----------------------------

__global__ __launch_bounds__(256) void cvt_f32_bf16(const float* __restrict__ in,
                                                    __bf16* __restrict__ out, int n) {
  int i = blockIdx.x * 256 + threadIdx.x;
  if (i < n) out[i] = f2bf(in[i]);
}

// in: [R][C] fp32 (row-major)  ->  out: [C][R] bf16 (row-major, i.e. transposed)
__global__ __launch_bounds__(256) void transpose_f32_bf16(const float* __restrict__ in,
                                                          __bf16* __restrict__ out,
                                                          int R, int C) {
  int i = blockIdx.x * 256 + threadIdx.x;
  if (i < R * C) {
    int r = i / C, c = i - r * C;
    out[(size_t)c * R + r] = f2bf(in[i]);
  }
}

// ------------------------------- WMMA GEMM ---------------------------------
// C[M,N] = A[M,K](bf16, row-major) x BT[N,K](bf16, row-major = B transposed)
// Block: 256 thr (8 waves), tile 128(M) x 64(N), K-step 32.
// LDS tiles staged by TDM (wave 0 issues descriptors; double-buffered against
// TENSORcnt) with a manual-staging fallback if the builtin is unavailable.
// Wave (wv&3, wv>>2) owns a 32x32 sub-tile => 2x2 v_wmma_f32_16x16x32_bf16.
// EPI==0: rotary + per-head scatter of q/k/v (QKV projection).
// EPI==1: +bias, fp32 store (output projection).

template <int EPI>
__global__ __launch_bounds__(256)
void gemm_wmma_bf16(const __bf16* __restrict__ A, const __bf16* __restrict__ BT,
                    int M, int N, int K,
                    const float* __restrict__ pos_emb,   // EPI==0
                    const float* __restrict__ b_out,     // EPI==1
                    __bf16* __restrict__ q_ws, __bf16* __restrict__ k_ws,
                    __bf16* __restrict__ v_ws,
                    float* __restrict__ out) {
  // +8 pad: 80B row stride (matches TDM pad fields), conflict-free frag reads
  __shared__ __bf16 As[2][128][40];
  __shared__ __bf16 Bs[2][64][40];

  const int tid  = threadIdx.x;
  const int lane = tid & 31;
  const int wv   = tid >> 5;
  const int m0   = blockIdx.y * 128;
  const int n0   = blockIdx.x * 64;
  const int mw   = (wv & 3) * 32;
  const int nw   = (wv >> 2) * 32;
  const bool lo  = (lane < 16);
  const int  l15 = lane & 15;

  v8f acc[2][2];
#pragma unroll
  for (int i = 0; i < 2; ++i)
#pragma unroll
    for (int j = 0; j < 2; ++j)
#pragma unroll
      for (int e = 0; e < 8; ++e) acc[i][j][e] = 0.0f;

  const int KSTEPS = K >> 5;

#if HAVE_TDM
  if (wv == 0) {  // one wave issues the DMA pair (EXEC-independent, per-wave op)
    tdm_load_2d((unsigned)(size_t)&As[0][0][0], A + (size_t)m0 * K,
                (unsigned)K, (unsigned)(M - m0), 32u, 128u, (unsigned)K);
    tdm_load_2d((unsigned)(size_t)&Bs[0][0][0], BT + (size_t)n0 * K,
                (unsigned)K, (unsigned)(N - n0), 32u, 64u, (unsigned)K);
  }
#endif

  for (int ks = 0; ks < KSTEPS; ++ks) {
    const int cur = ks & 1;
    const int k0  = ks * 32;

#if HAVE_TDM
    if (wv == 0) {
      if (ks + 1 < KSTEPS) {   // prefetch next tile pair into the other buffer
        const int k1 = k0 + 32;
        tdm_load_2d((unsigned)(size_t)&As[cur ^ 1][0][0], A + (size_t)m0 * K + k1,
                    (unsigned)(K - k1), (unsigned)(M - m0), 32u, 128u, (unsigned)K);
        tdm_load_2d((unsigned)(size_t)&Bs[cur ^ 1][0][0], BT + (size_t)n0 * K + k1,
                    (unsigned)(K - k1), (unsigned)(N - n0), 32u, 64u, (unsigned)K);
        __builtin_amdgcn_s_wait_tensorcnt(2);  // in-order: older pair is done
      } else {
        __builtin_amdgcn_s_wait_tensorcnt(0);
      }
    }
#else
    // ---- manual staging fallback ----
    {
      int row = tid >> 1, half = tid & 1;
      int gm  = m0 + row;
      int gmc = (gm < M) ? gm : (M - 1);
      const uint4* src = (const uint4*)(A + (size_t)gmc * K + k0) + half * 2;
      uint4 d0 = src[0], d1 = src[1];
      *(uint4*)&As[cur][row][half * 16]     = d0;
      *(uint4*)&As[cur][row][half * 16 + 8] = d1;
    }
    if (tid < 128) {
      int row = tid >> 1, half = tid & 1;
      const uint4* src = (const uint4*)(BT + (size_t)(n0 + row) * K + k0) + half * 2;
      uint4 d0 = src[0], d1 = src[1];
      *(uint4*)&Bs[cur][row][half * 16]     = d0;
      *(uint4*)&Bs[cur][row][half * 16 + 8] = d1;
    }
#endif
    __syncthreads();   // tile (TDM or manual) visible to all waves

    // ---- fragments (per CDNA5 16-bit A/B VGPR layouts) ----
    Frag af[2], bf[2];
    const int abase = lo ? 0 : 8;    // A: K 0..7|16..23 (lo) vs 8..15|24..31 (hi)
    const int bbase = lo ? 0 : 8;    // B: K 0..7|16..23 (lo) vs 8..15|24..31 (hi)
#pragma unroll
    for (int i = 0; i < 2; ++i) {
      int r = mw + l15 + i * 16;
      af[i].u[0] = *(const uint4*)&As[cur][r][abase];
      af[i].u[1] = *(const uint4*)&As[cur][r][abase + 16];
    }
#pragma unroll
    for (int j = 0; j < 2; ++j) {
      int r = nw + l15 + j * 16;
      bf[j].u[0] = *(const uint4*)&Bs[cur][r][bbase];
      bf[j].u[1] = *(const uint4*)&Bs[cur][r][bbase + 16];
    }
#pragma unroll
    for (int i = 0; i < 2; ++i)
#pragma unroll
      for (int j = 0; j < 2; ++j)
        acc[i][j] = wmma_bf16(af[i].v, bf[j].v, acc[i][j]);
    __syncthreads();   // everyone done reading before this buffer is re-filled
  }

  // ---- epilogue. C layout: VGPR p, lanes 0-15 -> row p, lanes 16-31 -> row p+8;
  //      col = lane&15. Rotary pair (2i,2i+1) lives in the adjacent lane.
#pragma unroll
  for (int i = 0; i < 2; ++i)
#pragma unroll
    for (int j = 0; j < 2; ++j)
#pragma unroll
      for (int p = 0; p < 8; ++p) {
        int gm = m0 + mw + i * 16 + p + (lo ? 0 : 8);
        int gc = n0 + nw + j * 16 + l15;
        float v = acc[i][j][p];
        if (EPI == 0) {
          float partner = __shfl_xor(v, 1, 32);     // uniform: outside the row guard
          int gms = (gm < M) ? gm : 0;
          int bb  = gms / NSEQ;
          int nn  = gms - bb * NSEQ;
          int s   = gc / DIM;                       // 0=q,1=k,2=v
          int rc  = gc - s * DIM;
          int hh  = rc >> 6;
          int dd  = rc & 63;
          float ov = v;
          if (s < 2) {                              // rotary on q,k
            float si = pos_emb[nn * DH + (dd >> 1)];
            float co = pos_emb[nn * DH + 32 + (dd >> 1)];
            ov = v * co + ((dd & 1) ? partner : -partner) * si;
          }
          if (gm < M) {
            __bf16* dst = (s == 0) ? q_ws : ((s == 1) ? k_ws : v_ws);
            dst[((size_t)(bb * HEADS + hh) * NSEQ + nn) * DH + dd] = f2bf(ov);
          }
        } else {
          if (gm < M) out[(size_t)gm * N + gc] = v + b_out[gc];
        }
      }
}

// ------------------------------- attention ---------------------------------
// One workgroup (128 thr / 4 waves) per (b,h). q/k/v are bf16 [b*h][n][64].
// Each wave owns a 16-row query tile: S row-block (16x197) fully in registers,
// mask+softmax via shfl across the 16-lane half, P restaged in LDS (C->A
// layout), V pre-transposed in LDS, then P@V with WMMA. Output stored as bf16
// in [b][n][h*dh] layout, ready to be the A operand of the out-proj GEMM.

__global__ __launch_bounds__(128)
void attention_wmma(const __bf16* __restrict__ q_ws, const __bf16* __restrict__ k_ws,
                    const __bf16* __restrict__ v_ws, const float* __restrict__ scale,
                    __bf16* __restrict__ attn_ws) {
  __shared__ __bf16 Vt[64][232];      // V transposed: Vt[d][j], j zero-padded to 224
  __shared__ __bf16 Ps[4][16][232];   // per-wave P staging (cols 208..231 zeroed)

  const int bh   = blockIdx.x;
  const int b    = bh / HEADS;
  const int h    = bh - b * HEADS;
  const int tid  = threadIdx.x;
  const int lane = tid & 31;
  const int wv   = tid >> 5;
  const bool lo  = (lane < 16);
  const int  l15 = lane & 15;

  const __bf16* qb = q_ws + (size_t)bh * NSEQ * DH;
  const __bf16* kb = k_ws + (size_t)bh * NSEQ * DH;
  const __bf16* vb = v_ws + (size_t)bh * NSEQ * DH;

  const __bf16 bz = f2bf(0.0f);
  for (int idx = tid; idx < 224 * 64; idx += 128) {   // coalesced read of V rows
    int j = idx >> 6, d = idx & 63;
    Vt[d][j] = (j < NSEQ) ? vb[j * DH + d] : bz;
  }
  for (int idx = tid; idx < 4 * 16 * 24; idx += 128) { // zero P pad cols 208..231
    int w = idx / (16 * 24);
    int r = (idx / 24) & 15;
    int c = 208 + (idx % 24);
    Ps[w][r][c] = bz;
  }
  __syncthreads();

  const float sc    = scale[h];
  const int   abase = lo ? 0 : 8;
  const int   bbase = lo ? 0 : 8;

  for (int rt = wv; rt < 13; rt += 4) {               // 13 row tiles of 16 (197)
    // --- Q fragments (A operand), rows rt*16 .. rt*16+15 ---
    Frag qf[2];
    {
      int r = rt * 16 + l15;
      if (r > NSEQ - 1) r = NSEQ - 1;                 // clamped rows never stored
      const __bf16* qrow = qb + (size_t)r * DH;
#pragma unroll
      for (int ks = 0; ks < 2; ++ks) {
        qf[ks].u[0] = *(const uint4*)(qrow + ks * 32 + abase);
        qf[ks].u[1] = *(const uint4*)(qrow + ks * 32 + abase + 16);
      }
    }
    // --- S = q.k^T for the whole row block (13 col tiles x 2 k-steps) ---
    v8f s[13];
#pragma unroll
    for (int jt = 0; jt < 13; ++jt) {
#pragma unroll
      for (int e = 0; e < 8; ++e) s[jt][e] = 0.0f;
      int j = jt * 16 + l15;
      if (j > NSEQ - 1) j = NSEQ - 1;
      const __bf16* krow = kb + (size_t)j * DH;       // B column j over d = K row j
#pragma unroll
      for (int ks = 0; ks < 2; ++ks) {
        Frag kf;
        kf.u[0] = *(const uint4*)(krow + ks * 32 + bbase);
        kf.u[1] = *(const uint4*)(krow + ks * 32 + bbase + 16);
        s[jt] = wmma_bf16(qf[ks].v, kf.v, s[jt]);
      }
    }
    // --- scale, diagonal + padding mask, softmax over j ---
    const int rbase = rt * 16 + (lo ? 0 : 8);
    float mx[8], sm[8];
#pragma unroll
    for (int p = 0; p < 8; ++p) mx[p] = NEG_BIG;
#pragma unroll
    for (int jt = 0; jt < 13; ++jt) {
      int col = jt * 16 + l15;
#pragma unroll
      for (int p = 0; p < 8; ++p) {
        float v = s[jt][p] * sc;
        int row = rbase + p;
        if (col == row) v = MASK_FILL;                // reference diag mask
        if (col > NSEQ - 1) v = NEG_BIG;              // tile padding
        s[jt][p] = v;
        mx[p] = fmaxf(mx[p], v);
      }
    }
#pragma unroll
    for (int off = 1; off < 16; off <<= 1)
#pragma unroll
      for (int p = 0; p < 8; ++p) mx[p] = fmaxf(mx[p], __shfl_xor(mx[p], off, 32));
#pragma unroll
    for (int p = 0; p < 8; ++p) sm[p] = 0.0f;
#pragma unroll
    for (int jt = 0; jt < 13; ++jt)
#pragma unroll
      for (int p = 0; p < 8; ++p) {
        float e = __expf(s[jt][p] - mx[p]);
        s[jt][p] = e;
        sm[p] += e;
      }
#pragma unroll
    for (int off = 1; off < 16; off <<= 1)
#pragma unroll
      for (int p = 0; p < 8; ++p) sm[p] += __shfl_xor(sm[p], off, 32);
#pragma unroll
    for (int p = 0; p < 8; ++p) sm[p] = 1.0f / sm[p];

    // --- restage P (C layout -> A layout) in this wave's LDS slice ---
#pragma unroll
    for (int jt = 0; jt < 13; ++jt) {
      int col = jt * 16 + l15;
#pragma unroll
      for (int p = 0; p < 8; ++p) {
        int rloc = p + (lo ? 0 : 8);
        Ps[wv][rloc][col] = f2bf(s[jt][p] * sm[p]);
      }
    }
    // (same-wave LDS RAW: DS ops are in order per wave; compiler inserts waits)

    // --- O = P @ V : 4 d-tiles x 7 k-steps of 32 ---
#pragma unroll
    for (int dt = 0; dt < 4; ++dt) {
      v8f o;
#pragma unroll
      for (int e = 0; e < 8; ++e) o[e] = 0.0f;
      int vrow = dt * 16 + l15;                       // B column d over j = Vt row d
#pragma unroll
      for (int js = 0; js < 7; ++js) {
        Frag pf, vf;
        pf.u[0] = *(const uint4*)&Ps[wv][l15][js * 32 + abase];
        pf.u[1] = *(const uint4*)&Ps[wv][l15][js * 32 + abase + 16];
        vf.u[0] = *(const uint4*)&Vt[vrow][js * 32 + bbase];
        vf.u[1] = *(const uint4*)&Vt[vrow][js * 32 + bbase + 16];
        o = wmma_bf16(pf.v, vf.v, o);
      }
#pragma unroll
      for (int p = 0; p < 8; ++p) {
        int n = rbase + p;
        if (n < NSEQ)
          attn_ws[((size_t)(b * NSEQ + n)) * DIM + h * DH + dt * 16 + l15] = f2bf(o[p]);
      }
    }
  }
}

// ------------------------------- launcher ----------------------------------

extern "C" void kernel_launch(void* const* d_in, const int* in_sizes, int n_in,
                              void* d_out, int out_size, void* d_ws, size_t ws_size,
                              hipStream_t stream) {
  const float* x      = (const float*)d_in[0];   // (64,197,768)
  const float* posemb = (const float*)d_in[1];   // (1,197,64)
  const float* w_qkv  = (const float*)d_in[2];   // (768,2304)
  const float* scale  = (const float*)d_in[3];   // (12,)
  const float* w_out  = (const float*)d_in[4];   // (768,768)
  const float* b_out  = (const float*)d_in[5];   // (768,)
  float*       out    = (float*)d_out;           // (64,197,768)

  char* ws = (char*)d_ws;
  size_t off = 0;
  __bf16* xb     = (__bf16*)(ws + off); off += (size_t)MROWS * DIM * 2;       // 19.4 MB
  __bf16* wqkvT  = (__bf16*)(ws + off); off += (size_t)NQKV * DIM * 2;        //  3.5 MB
  __bf16* woutT  = (__bf16*)(ws + off); off += (size_t)DIM * DIM * 2;         //  1.2 MB
  __bf16* q_ws   = (__bf16*)(ws + off); off += (size_t)BQ * HEADS * NSEQ * DH * 2;
  __bf16* k_ws   = (__bf16*)(ws + off); off += (size_t)BQ * HEADS * NSEQ * DH * 2;
  __bf16* v_ws   = (__bf16*)(ws + off); off += (size_t)BQ * HEADS * NSEQ * DH * 2;
  __bf16* attnws = (__bf16*)(ws + off); off += (size_t)MROWS * DIM * 2;       // ~102 MB total

  // precision conversion / weight transposition (one-time, bandwidth-trivial)
  cvt_f32_bf16<<<(MROWS * DIM + 255) / 256, 256, 0, stream>>>(x, xb, MROWS * DIM);
  transpose_f32_bf16<<<(DIM * NQKV + 255) / 256, 256, 0, stream>>>(w_qkv, wqkvT, DIM, NQKV);
  transpose_f32_bf16<<<(DIM * DIM + 255) / 256, 256, 0, stream>>>(w_out, woutT, DIM, DIM);

  // QKV projection + fused rotary + per-head scatter
  gemm_wmma_bf16<0><<<dim3(NQKV / 64, (MROWS + 127) / 128), 256, 0, stream>>>(
      xb, wqkvT, MROWS, NQKV, DIM, posemb, nullptr, q_ws, k_ws, v_ws, nullptr);

  // attention core: one block per (b,h)
  attention_wmma<<<BQ * HEADS, 128, 0, stream>>>(q_ws, k_ws, v_ws, scale, attnws);

  // output projection + bias
  gemm_wmma_bf16<1><<<dim3(DIM / 64, (MROWS + 127) / 128), 256, 0, stream>>>(
      attnws, woutT, MROWS, DIM, DIM, nullptr, b_out, nullptr, nullptr, nullptr, out);
}
